// PocModel_84928683311862
// MI455X (gfx1250) — compile-verified, matrix operand order checked by reading, not run on hardware
//
#include <hip/hip_runtime.h>
#include <math.h>

// ---------------- model dims ----------------
#define VOCAB 32000
#define DMODEL 1024
#define NEXP 8
#define DFF 2048
#define NLAYER 2
#define NTOK 8192          // B*T = 4*2048
#define LN_EPS 1e-5f

// ---------------- WMMA types ----------------
typedef __attribute__((ext_vector_type(16))) __bf16 v16bf;
typedef __attribute__((ext_vector_type(8)))  float  v8f;
typedef __attribute__((ext_vector_type(4)))  float  fvec4;   // clang-native, NT-capable

union FragU { uint4 q[2]; v16bf v; };

// f32 -> bf16 round-to-nearest-even (bit pattern)
__device__ __forceinline__ unsigned short f2bf(float f) {
    unsigned int u = __float_as_uint(f);
    u += 0x7fffu + ((u >> 16) & 1u);
    return (unsigned short)(u >> 16);
}

// Load one 16x32 bf16 A/B fragment (K-contiguous storage, leading dim `ld`
// in elements). Per the CDNA5 16-bit A layout: lanes 0-15 hold row M=lane,
// K[0:8] and K[16:24]; lanes 16-31 hold row M=lane-16, K[8:16] and K[24:32].
// Each chunk is 8 contiguous bf16 = one 16-byte load.
template <typename PT>
__device__ __forceinline__ v16bf load_frag(const PT* base, int ld, int lane) {
    int row = lane & 15;
    int k0  = (lane >> 4) << 3;          // 0 or 8
    const PT* p = base + (size_t)row * ld + k0;
    FragU u;
    u.q[0] = *reinterpret_cast<const uint4*>(p);        // K[k0 .. k0+7]
    u.q[1] = *reinterpret_cast<const uint4*>(p + 16);   // K[k0+16 .. k0+23]
    return u.v;
}

__device__ __forceinline__ v8f wmma_bf16(v16bf a, v16bf b, v8f c) {
    return __builtin_amdgcn_wmma_f32_16x16x32_bf16(
        /*neg_a=*/false, a, /*neg_b=*/false, b,
        /*c_mod=*/(short)0, c, /*reuse_a=*/false, /*reuse_b=*/false);
}

__device__ __forceinline__ float gelu_erf(float x) {
    return 0.5f * x * (1.0f + erff(x * 0.70710678118654752440f));
}

// ---------------- 1) tiled f32->bf16 transpose-convert ----------------
// in:  [R, C] f32 row-major (batched along z), out: [C, R] bf16 row-major.
// Source is read exactly once per call -> non-temporal loads; the bf16 copy
// is what we want resident in L2 (one layer's expert set = 67 MB < 192 MB).
__global__ void transpose_cvt(const float* __restrict__ in,
                              unsigned short* __restrict__ out,
                              int R, int C) {
    __shared__ float tile[32][33];
    size_t slice = (size_t)R * C;
    in  += (size_t)blockIdx.z * slice;
    out += (size_t)blockIdx.z * slice;
    int c0 = blockIdx.x * 32, r0 = blockIdx.y * 32;
#pragma unroll
    for (int j = 0; j < 4; ++j) {
        int r = r0 + threadIdx.y + j * 8;
        tile[threadIdx.y + j * 8][threadIdx.x] =
            __builtin_nontemporal_load(&in[(size_t)r * C + c0 + threadIdx.x]);
    }
    __syncthreads();
#pragma unroll
    for (int j = 0; j < 4; ++j) {
        int c = c0 + threadIdx.y + j * 8;
        out[(size_t)c * R + r0 + threadIdx.x] = f2bf(tile[threadIdx.x][threadIdx.y + j * 8]);
    }
}

// ---------------- 2) embedding gather ----------------
__global__ void gather_emb(const int* __restrict__ x,
                           const float* __restrict__ emb,
                           float* __restrict__ h) {
    int n = blockIdx.x;
    int tok = x[n];
    fvec4 v = __builtin_nontemporal_load(
        reinterpret_cast<const fvec4*>(emb + (size_t)tok * DMODEL) + threadIdx.x);
    reinterpret_cast<fvec4*>(h + (size_t)n * DMODEL)[threadIdx.x] = v;
}

// ---------------- 3) LayerNorm + gating softmax (one block / token) ------
__global__ void ln_gate(const float* __restrict__ h,
                        const float* __restrict__ g, const float* __restrict__ b,
                        const float* __restrict__ gw, const float* __restrict__ gb,
                        unsigned short* __restrict__ xnb,
                        float* __restrict__ rw) {
    __shared__ float sred[256];
    __shared__ float stats[2];
    __shared__ float slog[NEXP];
    int n = blockIdx.x, tid = threadIdx.x;
    int d0 = tid * 4;
    float4 hv = reinterpret_cast<const float4*>(h + (size_t)n * DMODEL)[tid];

    // mean
    sred[tid] = hv.x + hv.y + hv.z + hv.w;
    __syncthreads();
    for (int off = 128; off > 0; off >>= 1) {
        if (tid < off) sred[tid] += sred[tid + off];
        __syncthreads();
    }
    if (tid == 0) stats[0] = sred[0] * (1.0f / DMODEL);
    __syncthreads();
    float mu = stats[0];
    // variance
    float dx0 = hv.x - mu, dx1 = hv.y - mu, dx2 = hv.z - mu, dx3 = hv.w - mu;
    sred[tid] = dx0 * dx0 + dx1 * dx1 + dx2 * dx2 + dx3 * dx3;
    __syncthreads();
    for (int off = 128; off > 0; off >>= 1) {
        if (tid < off) sred[tid] += sred[tid + off];
        __syncthreads();
    }
    if (tid == 0) stats[1] = rsqrtf(sred[0] * (1.0f / DMODEL) + LN_EPS);
    __syncthreads();
    float rstd = stats[1];

    float xn0 = dx0 * rstd * g[d0 + 0] + b[d0 + 0];
    float xn1 = dx1 * rstd * g[d0 + 1] + b[d0 + 1];
    float xn2 = dx2 * rstd * g[d0 + 2] + b[d0 + 2];
    float xn3 = dx3 * rstd * g[d0 + 3] + b[d0 + 3];
    ushort4 o;
    o.x = f2bf(xn0); o.y = f2bf(xn1); o.z = f2bf(xn2); o.w = f2bf(xn3);
    reinterpret_cast<ushort4*>(xnb + (size_t)n * DMODEL)[tid] = o;

    // gating logits: xn @ gate_w  (gate_w is [D, E] row-major)
    float acc[NEXP];
#pragma unroll
    for (int e = 0; e < NEXP; ++e)
        acc[e] = xn0 * gw[(d0 + 0) * NEXP + e] + xn1 * gw[(d0 + 1) * NEXP + e] +
                 xn2 * gw[(d0 + 2) * NEXP + e] + xn3 * gw[(d0 + 3) * NEXP + e];
#pragma unroll
    for (int e = 0; e < NEXP; ++e) {
        sred[tid] = acc[e];
        __syncthreads();
        for (int off = 128; off > 0; off >>= 1) {
            if (tid < off) sred[tid] += sred[tid + off];
            __syncthreads();
        }
        if (tid == 0) slog[e] = sred[0] + gb[e];
        __syncthreads();
    }
    if (tid == 0) {
        float m = slog[0];
#pragma unroll
        for (int e = 1; e < NEXP; ++e) m = fmaxf(m, slog[e]);
        float s = 0.f, ex[NEXP];
#pragma unroll
        for (int e = 0; e < NEXP; ++e) { ex[e] = __expf(slog[e] - m); s += ex[e]; }
        float inv = 1.0f / s;
#pragma unroll
        for (int e = 0; e < NEXP; ++e) rw[(size_t)n * NEXP + e] = ex[e] * inv;
    }
}

// ---------------- 4) fused MoE expert kernel --------------------------
// One workgroup = 64-token tile, 8 waves. Loops experts; per expert:
//   phase A: act = gelu(xn_tile @ W1e)  -> LDS [64 x 2048] bf16 (256 KB)
//   phase B: eo  = act @ W2e            -> experts_out (NT), h += rw * eo
__global__ void moe_ffn(const unsigned short* __restrict__ xnb,
                        const unsigned short* __restrict__ w1T, // [E][DFF][D] bf16
                        const unsigned short* __restrict__ w2T, // [E][D][DFF] bf16
                        const float* __restrict__ rw,
                        float* __restrict__ h,
                        float* __restrict__ eo_out) {            // [N][E][D]
    extern __shared__ unsigned short lds_act[];                  // 64*2048
    const int wave = threadIdx.x >> 5;
    const int lane = threadIdx.x & 31;
    const int m_base = blockIdx.x * 64;
    const int mhi = (lane >> 4) << 3;     // row offset 0 or 8 in C layout
    const int nlo = lane & 15;

    for (int e = 0; e < NEXP; ++e) {
        const unsigned short* w1e = w1T + (size_t)e * DFF * DMODEL;
        const unsigned short* w2e = w2T + (size_t)e * DMODEL * DFF;

        // ---- phase A: 4 row-blocks x 32 col-groups (64 cols) = 128 jobs ----
        for (int job = wave; job < 128; job += 8) {
            int rb = job & 3;
            int i0 = (job >> 2) * 64;
            int m0 = m_base + rb * 16;
            v8f acc[4] = {};
            for (int k = 0; k < DMODEL; k += 32) {
                v16bf a = load_frag(xnb + (size_t)m0 * DMODEL + k, DMODEL, lane);
                __builtin_prefetch((const void*)(w1e + (size_t)i0 * DMODEL + k + 128), 0, 0);
#pragma unroll
                for (int t = 0; t < 4; ++t) {
                    v16bf bb = load_frag(w1e + (size_t)(i0 + t * 16) * DMODEL + k, DMODEL, lane);
                    acc[t] = wmma_bf16(a, bb, acc[t]);
                }
            }
#pragma unroll
            for (int t = 0; t < 4; ++t)
#pragma unroll
                for (int r = 0; r < 8; ++r) {
                    int m = rb * 16 + r + mhi;
                    int c = i0 + t * 16 + nlo;
                    lds_act[(size_t)m * DFF + c] = f2bf(gelu_erf(acc[t][r]));
                }
        }
        __syncthreads();

        // ---- phase B: 4 row-blocks x 16 col-groups (64 cols) = 64 jobs ----
        for (int job = wave; job < 64; job += 8) {
            int rb = job & 3;
            int c0 = (job >> 2) * 64;
            v8f acc[4] = {};
            for (int k = 0; k < DFF; k += 32) {
                v16bf a = load_frag(lds_act + (size_t)(rb * 16) * DFF + k, DFF, lane);
                __builtin_prefetch((const void*)(w2e + (size_t)c0 * DFF + k + 128), 0, 0);
#pragma unroll
                for (int t = 0; t < 4; ++t) {
                    v16bf bb = load_frag(w2e + (size_t)(c0 + t * 16) * DFF + k, DFF, lane);
                    acc[t] = wmma_bf16(a, bb, acc[t]);
                }
            }
#pragma unroll
            for (int t = 0; t < 4; ++t)
#pragma unroll
                for (int r = 0; r < 8; ++r) {
                    int tok = m_base + rb * 16 + r + mhi;
                    int c = c0 + t * 16 + nlo;
                    float v = acc[t][r];
                    // write-once 0.54 GB output stream: bypass caches
                    __builtin_nontemporal_store(
                        v, &eo_out[(size_t)tok * NEXP * DMODEL + (size_t)e * DMODEL + c]);
                    size_t hi = (size_t)tok * DMODEL + c;
                    h[hi] = h[hi] + rw[(size_t)tok * NEXP + e] * v;  // disjoint per wave
                }
        }
        __syncthreads();   // before next expert overwrites lds_act
    }
}

// ---------------- 5) h f32 -> bf16 ----------------
__global__ void cvt_f32_bf16(const float* __restrict__ in,
                             unsigned short* __restrict__ out, int n4) {
    int i = blockIdx.x * blockDim.x + threadIdx.x;
    int stride = gridDim.x * blockDim.x;
    for (; i < n4; i += stride) {
        float4 v = reinterpret_cast<const float4*>(in)[i];
        ushort4 o;
        o.x = f2bf(v.x); o.y = f2bf(v.y); o.z = f2bf(v.z); o.w = f2bf(v.w);
        reinterpret_cast<ushort4*>(out)[i] = o;
    }
}

// ---------------- 6) vocab head GEMM + bias ----------------
// C[N, V] = hb[N, D](bf16) x hwT[V, D](bf16)^T + bias.  Tile 64x128, 8 waves.
__global__ void head_gemm(const unsigned short* __restrict__ hb,
                          const unsigned short* __restrict__ hwT,
                          const float* __restrict__ bias,
                          float* __restrict__ out) {
    const int wave = threadIdx.x >> 5;
    const int lane = threadIdx.x & 31;
    const int rb = wave & 3, cg = wave >> 2;
    const int m0 = blockIdx.x * 64 + rb * 16;
    const int c0 = blockIdx.y * 128 + cg * 64;
    v8f acc[4] = {};
    for (int k = 0; k < DMODEL; k += 32) {
        v16bf a = load_frag(hb + (size_t)m0 * DMODEL + k, DMODEL, lane);
        __builtin_prefetch((const void*)(hwT + (size_t)c0 * DMODEL + k + 128), 0, 0);
#pragma unroll
        for (int t = 0; t < 4; ++t) {
            v16bf bb = load_frag(hwT + (size_t)(c0 + t * 16) * DMODEL + k, DMODEL, lane);
            acc[t] = wmma_bf16(a, bb, acc[t]);
        }
    }
    const int mhi = (lane >> 4) << 3;
    const int nlo = lane & 15;
#pragma unroll
    for (int t = 0; t < 4; ++t)
#pragma unroll
        for (int r = 0; r < 8; ++r) {
            int m = m0 + r + mhi;
            int n = c0 + t * 16 + nlo;
            // write-once 1.05 GB logits stream: bypass caches
            __builtin_nontemporal_store(acc[t][r] + bias[n], &out[(size_t)m * VOCAB + n]);
        }
}

// ---------------- host launcher ----------------
extern "C" void kernel_launch(void* const* d_in, const int* in_sizes, int n_in,
                              void* d_out, int out_size, void* d_ws, size_t ws_size,
                              hipStream_t stream) {
    const int*   x      = (const int*)  d_in[0];
    const float* emb    = (const float*)d_in[1];
    const float* ln_g   = (const float*)d_in[2];
    const float* ln_b   = (const float*)d_in[3];
    const float* gate_w = (const float*)d_in[4];
    const float* gate_b = (const float*)d_in[5];
    const float* w1     = (const float*)d_in[6];   // [L,E,D,I]
    const float* w2     = (const float*)d_in[7];   // [L,E,I,D]
    const float* head_w = (const float*)d_in[8];   // [D,V]
    const float* head_b = (const float*)d_in[9];

    float* logits  = (float*)d_out;                              // [N,V]
    float* experts = (float*)d_out + (size_t)NTOK * VOCAB;       // [L,N,E,D]

    // workspace carve (all 256B aligned by construction)
    char* p = (char*)d_ws;
    float* h            = (float*)p;          p += (size_t)NTOK * DMODEL * 4;
    unsigned short* xnb = (unsigned short*)p; p += (size_t)NTOK * DMODEL * 2;
    float* rw           = (float*)p;          p += (size_t)NTOK * NEXP * 4;
    unsigned short* w1T = (unsigned short*)p; p += (size_t)NLAYER * NEXP * DFF * DMODEL * 2;
    unsigned short* w2T = (unsigned short*)p; p += (size_t)NLAYER * NEXP * DMODEL * DFF * 2;
    unsigned short* hwT = (unsigned short*)p; p += (size_t)VOCAB * DMODEL * 2;
    (void)ws_size; (void)in_sizes; (void)n_in; (void)out_size;

    dim3 tblk(32, 8);
    // W1 [D,I] -> [I,D] bf16, batched over L*E
    transpose_cvt<<<dim3(DFF / 32, DMODEL / 32, NLAYER * NEXP), tblk, 0, stream>>>(
        w1, w1T, DMODEL, DFF);
    // W2 [I,D] -> [D,I] bf16
    transpose_cvt<<<dim3(DMODEL / 32, DFF / 32, NLAYER * NEXP), tblk, 0, stream>>>(
        w2, w2T, DFF, DMODEL);
    // head_w [D,V] -> [V,D] bf16
    transpose_cvt<<<dim3(VOCAB / 32, DMODEL / 32, 1), tblk, 0, stream>>>(
        head_w, hwT, DMODEL, VOCAB);

    gather_emb<<<NTOK, 256, 0, stream>>>(x, emb, h);

    for (int l = 0; l < NLAYER; ++l) {
        ln_gate<<<NTOK, 256, 0, stream>>>(
            h, ln_g + (size_t)l * DMODEL, ln_b + (size_t)l * DMODEL,
            gate_w + (size_t)l * DMODEL * NEXP, gate_b + (size_t)l * NEXP,
            xnb, rw);
        moe_ffn<<<NTOK / 64, 256, 64 * DFF * sizeof(unsigned short), stream>>>(
            xnb,
            w1T + (size_t)l * NEXP * DFF * DMODEL,
            w2T + (size_t)l * NEXP * DMODEL * DFF,
            rw, h,
            experts + (size_t)l * NTOK * NEXP * DMODEL);
    }

    cvt_f32_bf16<<<2048, 256, 0, stream>>>(h, xnb, NTOK * DMODEL / 4);
    head_gemm<<<dim3(NTOK / 64, VOCAB / 128), 256, 0, stream>>>(xnb, hwT, head_b, logits);
}
